// RoPECrossAttention_87969520157411
// MI455X (gfx1250) — matrix operand order repaired and checked by reading, not compiled, not run on hardware
//
#include <hip/hip_runtime.h>

// ---------------------------------------------------------------------------
// RoPE cross-attention, CDNA5 (gfx1250) wave32 WMMA implementation.
// B=4, NQ=1024, NKV=2048, D=1024, H=16, HD=64.
// ---------------------------------------------------------------------------

typedef _Float16 v16h __attribute__((ext_vector_type(16)));
typedef float    v8f  __attribute__((ext_vector_type(8)));

union AF {                 // one WMMA operand fragment (16 halves per lane)
  v16h v;
  uint4 u[2];
  _Float16 h[16];
};

__device__ __forceinline__ v8f wmma16(v16h a, v16h b, v8f c) {
  return __builtin_amdgcn_wmma_f32_16x16x32_f16(false, a, false, b,
                                                (short)0, c, false, false);
}

constexpr int kB   = 4;
constexpr int kNQ  = 1024;
constexpr int kNKV = 2048;
constexpr int kD   = 1024;
constexpr int kH   = 16;
constexpr int kHD  = 64;

// ---------------------------------------------------------------------------
// LayerNorm (per row of 1024) -> f16
// ---------------------------------------------------------------------------
__global__ __launch_bounds__(256) void ln_rows(const float* __restrict__ x,
                                               _Float16* __restrict__ y) {
  __shared__ float red[256];
  const int row = blockIdx.x;
  const int t = threadIdx.x;
  const float* xr = x + (size_t)row * kD;
  float v[4];
#pragma unroll
  for (int i = 0; i < 4; ++i) v[i] = xr[t + 256 * i];
  float s = v[0] + v[1] + v[2] + v[3];
  red[t] = s;
  __syncthreads();
  for (int off = 128; off > 0; off >>= 1) {
    if (t < off) red[t] += red[t + off];
    __syncthreads();
  }
  const float mu = red[0] * (1.0f / kD);
  __syncthreads();
  float ss = 0.f;
#pragma unroll
  for (int i = 0; i < 4; ++i) {
    float d = v[i] - mu;
    ss += d * d;
  }
  red[t] = ss;
  __syncthreads();
  for (int off = 128; off > 0; off >>= 1) {
    if (t < off) red[t] += red[t + off];
    __syncthreads();
  }
  const float rs = rsqrtf(red[0] * (1.0f / kD) + 1e-5f);
  _Float16* yr = y + (size_t)row * kD;
#pragma unroll
  for (int i = 0; i < 4; ++i) yr[t + 256 * i] = (_Float16)((v[i] - mu) * rs);
}

// ---------------------------------------------------------------------------
// f32 -> f16 cast (weights)
// ---------------------------------------------------------------------------
__global__ __launch_bounds__(256) void cast_f2h(const float* __restrict__ s,
                                                _Float16* __restrict__ d, int n) {
  int i = blockIdx.x * 256 + threadIdx.x;
  if (i < n) d[i] = (_Float16)s[i];
}

// ---------------------------------------------------------------------------
// GEMM: Y = X @ W^T + bias, optional RoPE, three output layouts.
// X: [rows][1024] f16 row-major.  W: [1024][1024] f16 row-major (so B-matrix
// column n == W row n, contiguous).  Block = 128 thr (4 waves), tile 64x64:
// wave w computes rows m0+w*16..+15 over 64 cols (== one head: n0=64*blockIdx.y).
//   OUTMODE=0: f16 head-major [B][H][rowsPerBatch][64]      (+RoPE if ROPE)
//   OUTMODE=1: f32 row-major [rows][1024]                   (final projection)
//   OUTMODE=2: f16 head-major transposed [B][H][64][rowsPerBatch]  (V)
// ---------------------------------------------------------------------------
template <int ROPE, int OUTMODE>
__global__ __launch_bounds__(128) void gemm_proj(
    const _Float16* __restrict__ X, const _Float16* __restrict__ W,
    const float* __restrict__ bias, const float* __restrict__ cs,
    const float* __restrict__ sn, _Float16* __restrict__ outh,
    float* __restrict__ outf, int rowsPerBatch) {
  __shared__ float tile[64 * 64];
  const int tid = threadIdx.x;
  const int wave = tid >> 5;
  const int lane = tid & 31;
  const int ln16 = lane & 15;
  const int hi = lane >> 4;
  const int m0 = blockIdx.x * 64 + wave * 16;
  const int n0 = blockIdx.y * 64;
  const _Float16* xrow = X + (size_t)(m0 + ln16) * kD;

  v8f c[4] = {};
  for (int kk = 0; kk < kD; kk += 32) {
    AF a;
    const _Float16* p = xrow + kk + hi * 16;
    a.u[0] = *reinterpret_cast<const uint4*>(p);       // K = kk+16*hi .. +7
    a.u[1] = *reinterpret_cast<const uint4*>(p + 8);   // K = kk+16*hi+8 .. +7
#pragma unroll
    for (int j = 0; j < 4; ++j) {
      AF b;
      const _Float16* wp =
          W + (size_t)(n0 + 16 * j + ln16) * kD + kk + hi * 16;
      b.u[0] = *reinterpret_cast<const uint4*>(wp);
      b.u[1] = *reinterpret_cast<const uint4*>(wp + 8);
      c[j] = wmma16(a.v, b.v, c[j]);
    }
  }

  // bias + stage f32 tile to LDS (C layout: VGPR r, lane -> M=r+8*hi, N=ln16)
#pragma unroll
  for (int j = 0; j < 4; ++j) {
    const float bv = bias[n0 + 16 * j + ln16];
#pragma unroll
    for (int r = 0; r < 8; ++r) {
      tile[(wave * 16 + r + 8 * hi) * 64 + 16 * j + ln16] = c[j][r] + bv;
    }
  }
  __syncthreads();

  for (int idx = tid; idx < 64 * 64; idx += 128) {
    const int rl = idx >> 6;
    const int d = idx & 63;
    const int grow = blockIdx.x * 64 + rl;
    float y = tile[idx];
    if (ROPE) {
      const int q = grow % rowsPerBatch;  // sequence position
      const float rot =
          (d < 32) ? -tile[rl * 64 + d + 32] : tile[rl * 64 + d - 32];
      y = y * cs[q * kHD + d] + rot * sn[q * kHD + d];
    }
    if (OUTMODE == 1) {
      outf[(size_t)grow * kD + n0 + d] = y;
    } else {
      const int b = grow / rowsPerBatch;
      const int q = grow % rowsPerBatch;
      const int h = blockIdx.y;  // n0/64 == head index
      if (OUTMODE == 0) {
        outh[((size_t)(b * kH + h) * rowsPerBatch + q) * kHD + d] =
            (_Float16)y;
      } else {  // OUTMODE == 2: transposed head-major (for V)
        outh[((size_t)(b * kH + h) * kHD + d) * rowsPerBatch + q] =
            (_Float16)y;
      }
    }
  }
}

// ---------------------------------------------------------------------------
// Flash attention.  Q/K head-major f16 [B][H][N][64]; V transposed head-major
// f16 [B][H][64][NKV] so P@V B-fragments are contiguous 16-half loads.
// Grid: (B*H, NQ/64), block 128.  Each wave owns 16 query rows; KV tiles of
// 64; online softmax; P staged in per-wave LDS region and re-read in
// A-fragment layout.  Output: f16 row-major [B][NQ][1024].
// ---------------------------------------------------------------------------
__global__ __launch_bounds__(128) void attention(const _Float16* __restrict__ Q,
                                                 const _Float16* __restrict__ K,
                                                 const _Float16* __restrict__ Vt,
                                                 _Float16* __restrict__ ao) {
  __shared__ _Float16 pshm[4][16 * 64];
  const int tid = threadIdx.x;
  const int wave = tid >> 5;
  const int lane = tid & 31;
  const int ln16 = lane & 15;
  const int hi = lane >> 4;
  const int bh = blockIdx.x;
  const int b = bh >> 4;
  const int h = bh & 15;
  const int q0 = blockIdx.y * 64 + wave * 16;

  // Q fragments for this wave's 16 rows, scale 1/sqrt(64)=0.125 folded in.
  AF qa[2];
  {
    const _Float16* qrow = Q + ((size_t)bh * kNQ + q0 + ln16) * kHD;
#pragma unroll
    for (int f = 0; f < 2; ++f) {
      const _Float16* p = qrow + 32 * f + hi * 16;
      qa[f].u[0] = *reinterpret_cast<const uint4*>(p);
      qa[f].u[1] = *reinterpret_cast<const uint4*>(p + 8);
#pragma unroll
      for (int i = 0; i < 16; ++i) qa[f].h[i] = qa[f].h[i] * (_Float16)0.125f;
    }
  }

  const _Float16* Kb = K + (size_t)bh * kNKV * kHD;   // [n][64]
  const _Float16* Vb = Vt + (size_t)bh * kHD * kNKV;  // [d][NKV]

  v8f o[4] = {};
  float mrun[8], lrun[8];
#pragma unroll
  for (int r = 0; r < 8; ++r) {
    mrun[r] = -3.0e38f;
    lrun[r] = 0.f;
  }

  for (int nt = 0; nt < kNKV; nt += 64) {
    // Prefetch next KV tile toward L2/WGP$ (global_prefetch_b8).
    if (nt + 64 < kNKV) {
      __builtin_prefetch(Kb + (size_t)(nt + 64 + lane * 2) * kHD, 0, 1);
      __builtin_prefetch(Vb + (size_t)(lane * 2) * kNKV + nt + 64, 0, 1);
    }

    // ---- S = (Q*scale) @ K^T : 4 C-tiles of 16x16 -------------------------
    v8f s[4];
#pragma unroll
    for (int j = 0; j < 4; ++j) {
      const _Float16* kp = Kb + (size_t)(nt + 16 * j + ln16) * kHD + hi * 16;
      AF bk;
      bk.u[0] = *reinterpret_cast<const uint4*>(kp);
      bk.u[1] = *reinterpret_cast<const uint4*>(kp + 8);
      v8f acc = {};
      acc = wmma16(qa[0].v, bk.v, acc);
      AF bk2;
      bk2.u[0] = *reinterpret_cast<const uint4*>(kp + 32);
      bk2.u[1] = *reinterpret_cast<const uint4*>(kp + 40);
      acc = wmma16(qa[1].v, bk2.v, acc);
      s[j] = acc;
    }

    // ---- online softmax row stats (rows split across 16-lane halves) ------
    float pm[8];
#pragma unroll
    for (int r = 0; r < 8; ++r) {
      float t = s[0][r];
#pragma unroll
      for (int j = 1; j < 4; ++j) t = fmaxf(t, s[j][r]);
#pragma unroll
      for (int off = 1; off < 16; off <<= 1) t = fmaxf(t, __shfl_xor(t, off, 16));
      pm[r] = t;
    }
    float alpha[8], mnew[8];
#pragma unroll
    for (int r = 0; r < 8; ++r) {
      mnew[r] = fmaxf(mrun[r], pm[r]);
      alpha[r] = __expf(mrun[r] - mnew[r]);
      mrun[r] = mnew[r];
    }
    float psum[8];
#pragma unroll
    for (int r = 0; r < 8; ++r) psum[r] = 0.f;
#pragma unroll
    for (int j = 0; j < 4; ++j) {
#pragma unroll
      for (int r = 0; r < 8; ++r) {
        const float p = __expf(s[j][r] - mnew[r]);
        psum[r] += p;
        pshm[wave][(r + 8 * hi) * 64 + 16 * j + ln16] = (_Float16)p;
      }
    }
#pragma unroll
    for (int r = 0; r < 8; ++r) {
      float t = psum[r];
#pragma unroll
      for (int off = 1; off < 16; off <<= 1) t += __shfl_xor(t, off, 16);
      lrun[r] = lrun[r] * alpha[r] + t;
    }
#pragma unroll
    for (int j = 0; j < 4; ++j)
#pragma unroll
      for (int r = 0; r < 8; ++r) o[j][r] *= alpha[r];

    __syncthreads();  // P stores visible before A-layout reloads

    // ---- reload P as A fragments, accumulate O += P @ V -------------------
    AF pa[2];
#pragma unroll
    for (int f = 0; f < 2; ++f) {
      const _Float16* base = &pshm[wave][ln16 * 64 + 32 * f + hi * 16];
      pa[f].u[0] = *reinterpret_cast<const uint4*>(base);
      pa[f].u[1] = *reinterpret_cast<const uint4*>(base + 8);
    }
#pragma unroll
    for (int jd = 0; jd < 4; ++jd) {
#pragma unroll
      for (int f = 0; f < 2; ++f) {
        // B[n][d] column d == Vt row d: contiguous 16 halves at n-offset.
        const _Float16* vp =
            Vb + (size_t)(16 * jd + ln16) * kNKV + nt + 32 * f + hi * 16;
        AF bv;
        bv.u[0] = *reinterpret_cast<const uint4*>(vp);
        bv.u[1] = *reinterpret_cast<const uint4*>(vp + 8);
        o[jd] = wmma16(pa[f].v, bv.v, o[jd]);
      }
    }
    __syncthreads();
  }

  // ---- normalize and store row-major f16 [B][NQ][1024] ---------------------
  float inv[8];
#pragma unroll
  for (int r = 0; r < 8; ++r) inv[r] = 1.0f / lrun[r];
#pragma unroll
  for (int jd = 0; jd < 4; ++jd) {
#pragma unroll
    for (int r = 0; r < 8; ++r) {
      const int q = q0 + r + 8 * hi;
      ao[((size_t)(b * kNQ + q)) * kD + h * kHD + 16 * jd + ln16] =
          (_Float16)(o[jd][r] * inv[r]);
    }
  }
}

// ---------------------------------------------------------------------------
// Host launch
// ---------------------------------------------------------------------------
extern "C" void kernel_launch(void* const* d_in, const int* in_sizes, int n_in,
                              void* d_out, int out_size, void* d_ws,
                              size_t ws_size, hipStream_t stream) {
  const float* q_in = (const float*)d_in[0];
  const float* kv_in = (const float*)d_in[1];
  const float* q_cos = (const float*)d_in[2];
  const float* q_sin = (const float*)d_in[3];
  const float* kv_cos = (const float*)d_in[4];
  const float* kv_sin = (const float*)d_in[5];
  const float* Wq = (const float*)d_in[6];
  const float* bq = (const float*)d_in[7];
  const float* Wk = (const float*)d_in[8];
  const float* bk = (const float*)d_in[9];
  const float* Wv = (const float*)d_in[10];
  const float* bv = (const float*)d_in[11];
  const float* Wo = (const float*)d_in[12];
  const float* bo = (const float*)d_in[13];
  float* out = (float*)d_out;

  char* base = (char*)d_ws;
  size_t off = 0;
  auto alloc = [&](size_t bytes) {
    void* p = base + off;
    off = (off + bytes + 255) & ~(size_t)255;
    return p;
  };
  const size_t rowsQ = (size_t)kB * kNQ;    // 4096
  const size_t rowsKV = (size_t)kB * kNKV;  // 8192

  _Float16* qln = (_Float16*)alloc(rowsQ * kD * 2);
  _Float16* kvln = (_Float16*)alloc(rowsKV * kD * 2);
  _Float16* wqh = (_Float16*)alloc((size_t)kD * kD * 2);
  _Float16* wkh = (_Float16*)alloc((size_t)kD * kD * 2);
  _Float16* wvh = (_Float16*)alloc((size_t)kD * kD * 2);
  _Float16* woh = (_Float16*)alloc((size_t)kD * kD * 2);
  _Float16* Qh = (_Float16*)alloc(rowsQ * kD * 2);   // [B][H][NQ][64]
  _Float16* Kh = (_Float16*)alloc(rowsKV * kD * 2);  // [B][H][NKV][64]
  _Float16* Vh = (_Float16*)alloc(rowsKV * kD * 2);  // [B][H][64][NKV]
  _Float16* ao = (_Float16*)alloc(rowsQ * kD * 2);   // [B][NQ][1024]

  // 1) LayerNorm -> f16
  ln_rows<<<(int)rowsQ, 256, 0, stream>>>(q_in, qln);
  ln_rows<<<(int)rowsKV, 256, 0, stream>>>(kv_in, kvln);

  // 2) weights -> f16
  const int wn = kD * kD;
  cast_f2h<<<(wn + 255) / 256, 256, 0, stream>>>(Wq, wqh, wn);
  cast_f2h<<<(wn + 255) / 256, 256, 0, stream>>>(Wk, wkh, wn);
  cast_f2h<<<(wn + 255) / 256, 256, 0, stream>>>(Wv, wvh, wn);
  cast_f2h<<<(wn + 255) / 256, 256, 0, stream>>>(Wo, woh, wn);

  // 3) projections (+RoPE for Q/K), head-major f16 outputs (V transposed)
  gemm_proj<1, 0><<<dim3(rowsQ / 64, kD / 64), 128, 0, stream>>>(
      qln, wqh, bq, q_cos, q_sin, Qh, nullptr, kNQ);
  gemm_proj<1, 0><<<dim3(rowsKV / 64, kD / 64), 128, 0, stream>>>(
      kvln, wkh, bk, kv_cos, kv_sin, Kh, nullptr, kNKV);
  gemm_proj<0, 2><<<dim3(rowsKV / 64, kD / 64), 128, 0, stream>>>(
      kvln, wvh, bv, nullptr, nullptr, Vh, nullptr, kNKV);

  // 4) flash attention -> row-major f16
  attention<<<dim3(kB * kH, kNQ / 64), 128, 0, stream>>>(Qh, Kh, Vh, ao);

  // 5) output projection -> f32 d_out
  gemm_proj<0, 1><<<dim3(rowsQ / 64, kD / 64), 128, 0, stream>>>(
      ao, woh, bo, nullptr, nullptr, nullptr, out, kNQ);
}
